// WGCN_26809185861706
// MI455X (gfx1250) — matrix operand order repaired and checked by reading, not compiled, not run on hardware
//
#include <hip/hip_runtime.h>
#include <hip/hip_bf16.h>

// ---------------------------------------------------------------------------
// Weighted 3-layer GCN for MI455X (gfx1250, wave32).
// GEMMs run on the matrix pipes via v_wmma_f32_16x16x32_f16 (f16 in, f32 acc),
// 16x32 output per wave (A-operand reused across two WMMAs per K-step).
// Edge aggregation is one-wave-per-edge coalesced gather + native f32 L2
// atomics. Node table (25.6MB) is L2-resident (192MB), so the scatter/gather
// phase is L2-bandwidth bound, not HBM bound.
// ---------------------------------------------------------------------------

typedef __attribute__((ext_vector_type(16))) _Float16 v16h;
typedef __attribute__((ext_vector_type(8)))  _Float16 v8h;
typedef __attribute__((ext_vector_type(8)))  float    v8f;

#define DH 128  // d_in == d_h == 128 in the reference

static __device__ __forceinline__ void atomic_add_f32(float* p, float v) {
  __hip_atomic_fetch_add(p, v, __ATOMIC_RELAXED, __HIP_MEMORY_SCOPE_AGENT);
}

// ---- utility ---------------------------------------------------------------
__global__ void k_zero(float* __restrict__ p, int n) {
  int t = blockIdx.x * blockDim.x + threadIdx.x;
  if (t < n) p[t] = 0.0f;
}

// ---- degree / weight-degree accumulation (segment sums via L2 atomics) -----
__global__ void k_degrees(const int* __restrict__ src, const int* __restrict__ dst,
                          const float* __restrict__ ew,
                          float* __restrict__ ow, float* __restrict__ iw,
                          float* __restrict__ od, float* __restrict__ id, int E) {
  int e = blockIdx.x * blockDim.x + threadIdx.x;
  if (e >= E) return;
  float w = ew[e];
  int s = src[e], t = dst[e];
  atomic_add_f32(&ow[s], w);
  atomic_add_f32(&iw[t], w);
  atomic_add_f32(&od[s], 1.0f);
  atomic_add_f32(&id[t], 1.0f);
}

// In-place: weighted degrees -> inv-sqrt (0 if deg<=0); unweighted -> clipped inv-sqrt
__global__ void k_node_norm(float* __restrict__ ow, float* __restrict__ iw,
                            float* __restrict__ od, float* __restrict__ id, int N) {
  int i = blockIdx.x * blockDim.x + threadIdx.x;
  if (i >= N) return;
  float a = ow[i]; ow[i] = (a > 0.0f) ? rsqrtf(a) : 0.0f;
  float b = iw[i]; iw[i] = (b > 0.0f) ? rsqrtf(b) : 0.0f;
  od[i] = rsqrtf(fmaxf(od[i], 1.0f));
  id[i] = rsqrtf(fmaxf(id[i], 1.0f));
}

// new_w[e] = ew * out_wdeg(src)^-1/2 * in_wdeg(dst)^-1/2
__global__ void k_edge_norm(const int* __restrict__ src, const int* __restrict__ dst,
                            const float* __restrict__ ew,
                            const float* __restrict__ ow, const float* __restrict__ iw,
                            float* __restrict__ nw, int E) {
  int e = blockIdx.x * blockDim.x + threadIdx.x;
  if (e >= E) return;
  nw[e] = ew[e] * ow[src[e]] * iw[dst[e]];
}

// Xh[i,k] = f16( X[i,k] * out_deg_invsqrt[i] )   (rows of width DH=128)
__global__ void k_scale_cvt(const float* __restrict__ X, const float* __restrict__ odi,
                            _Float16* __restrict__ Xh, int total) {
  int t = blockIdx.x * blockDim.x + threadIdx.x;
  if (t >= total) return;
  int row = t >> 7;  // /128
  Xh[t] = (_Float16)(X[t] * odi[row]);
}

// WhT[n,k] = f16( W[k,n] ),  W is [128 x dout] row-major
__global__ void k_transpose_cvt(const float* __restrict__ W, _Float16* __restrict__ WhT,
                                int dout) {
  int t = blockIdx.x * blockDim.x + threadIdx.x;
  if (t >= DH * dout) return;
  int k = t / dout, n = t - k * dout;
  WhT[n * DH + k] = (_Float16)W[t];
}

// ---- WMMA GEMM: Y[Mpad x dout] = Xh[Mpad x 128] @ Wh[128 x dout] -----------
// One wave computes a 16 x (16*NT) output block; K=128 as 4 steps of
// v_wmma_f32_16x16x32_f16; the A operand registers are reused across the NT
// WMMAs of each K-step (halves A-side load traffic for NT=2).
// A layout (ISA 7.12.2, 16-bit A 16x32): lane l -> row l%16,
//   halves 0..7 = K (l/16)*8..+7, halves 8..15 = same +16.
// B layout (documented 16-bit B striping): lane l -> col l%16,
//   halves 0..7 = K (l/16)*8..+7, halves 8..15 = same +16 (read from WhT[n][k]).
// Output rows are padded to a multiple of 16 so stores need no guards.
template <int NT>
__global__ __launch_bounds__(32)
void k_gemm_wmma(const _Float16* __restrict__ Xh, const _Float16* __restrict__ WhT,
                 float* __restrict__ Y, int dout) {
  const int lane  = threadIdx.x;      // 0..31, EXEC all 1s (WMMA requirement)
  const int sub   = lane & 15;
  const int sel   = lane >> 4;        // 0 or 1
  const int mbase = blockIdx.x * 16;
  const int nbase = blockIdx.y * (16 * NT);

  const _Float16* arow = Xh + (mbase + sub) * DH;

  v8f c[NT];
#pragma unroll
  for (int t = 0; t < NT; ++t) c[t] = (v8f){};

#pragma unroll
  for (int kt = 0; kt < 4; ++kt) {
    const int kb = kt * 32;
    v16h a;
    *((v8h*)&a)     = *(const v8h*)(arow + kb + sel * 8);       // K lo half
    *((v8h*)&a + 1) = *(const v8h*)(arow + kb + 16 + sel * 8);  // K hi half
#pragma unroll
    for (int t = 0; t < NT; ++t) {
      const _Float16* bcol = WhT + (nbase + t * 16 + sub) * DH;
      v16h b;
      *((v8h*)&b)     = *(const v8h*)(bcol + kb + sel * 8);       // K 0..7 / 8..15
      *((v8h*)&b + 1) = *(const v8h*)(bcol + kb + 16 + sel * 8);  // K 16..23 / 24..31
      c[t] = __builtin_amdgcn_wmma_f32_16x16x32_f16(
          /*neg_a=*/false, a, /*neg_b=*/false, b,
          /*c_mod=*/(short)0, c[t], /*reuse_a=*/false, /*reuse_b=*/false);
    }
  }

  // C/D layout: VGPR v -> row (v + 8*sel), col = lane%16. Unguarded stores:
  // Y has padded rows, so the tail tile writes land in scratch padding.
  const int r0 = mbase + sel * 8;
#pragma unroll
  for (int t = 0; t < NT; ++t) {
    float* yp = Y + nbase + t * 16 + sub;
#pragma unroll
    for (int v = 0; v < 8; ++v)
      yp[(r0 + v) * dout] = c[t][v];
  }
}

// ---- edge aggregation: agg[dst] += Y[src] * nw[e]  (one wave per edge) -----
__global__ void k_edge_agg(const float* __restrict__ Y, const float* __restrict__ nw,
                           const int* __restrict__ src, const int* __restrict__ dst,
                           float* __restrict__ agg, int E, int d) {
  int gtid = blockIdx.x * blockDim.x + threadIdx.x;
  int e    = gtid >> 5;
  int lane = gtid & 31;
  if (e >= E) return;
  const float w = nw[e];
  const float* yr = Y + (long)src[e] * d;
  float* ar = agg + (long)dst[e] * d;
  for (int j = lane; j < d; j += 32)           // coalesced b32 loads + f32 atomics
    atomic_add_f32(&ar[j], yr[j] * w);
}

// ---- out = act( agg * in_deg^-1/2 + bias ) ---------------------------------
__global__ void k_finalize(const float* __restrict__ agg, const float* __restrict__ idi,
                           const float* __restrict__ bias, float* __restrict__ out,
                           int N, int d, int relu) {
  int t = blockIdx.x * blockDim.x + threadIdx.x;
  if (t >= N * d) return;
  int i = t / d, j = t - i * d;
  float v = agg[t] * idi[i] + bias[j];
  out[t] = relu ? fmaxf(v, 0.0f) : v;
}

// ---------------------------------------------------------------------------
extern "C" void kernel_launch(void* const* d_in, const int* in_sizes, int n_in,
                              void* d_out, int out_size, void* d_ws, size_t ws_size,
                              hipStream_t stream) {
  const float* features = (const float*)d_in[0];
  const int*   src      = (const int*)  d_in[1];
  const int*   dst      = (const int*)  d_in[2];
  const float* ew       = (const float*)d_in[3];
  const float* W0       = (const float*)d_in[4];
  const float* b0       = (const float*)d_in[5];
  const float* W1       = (const float*)d_in[6];
  const float* b1       = (const float*)d_in[7];
  const float* W2       = (const float*)d_in[8];
  const float* b2       = (const float*)d_in[9];

  const int N    = in_sizes[0] / DH;   // 50000
  const int E    = in_sizes[1];        // 640000
  const int NCLS = in_sizes[9];        // 16
  const int Mpad = (N + 15) & ~15;     // padded rows: guard-free WMMA loads/stores

  // --- carve workspace (256B-aligned regions), ~67MB total ---
  char*  ws  = (char*)d_ws;
  size_t off = 0;
  auto carve = [&](size_t bytes) -> void* {
    void* p = ws + off;
    off = (off + bytes + 255) & ~(size_t)255;
    return p;
  };
  const int Npad = (N + 63) & ~63;
  float*    stats = (float*)   carve((size_t)4 * Npad * 4); // ow|iw|od|id contiguous
  float*    ow = stats, *iw = stats + Npad, *od = stats + 2 * Npad, *id = stats + 3 * Npad;
  float*    nw    = (float*)   carve((size_t)E * 4);
  _Float16* Xh    = (_Float16*)carve((size_t)Mpad * DH * 2);
  _Float16* WhT   = (_Float16*)carve((size_t)DH * DH * 2);
  float*    Y     = (float*)   carve((size_t)Mpad * DH * 4); // GEMM out / next act
  float*    agg   = (float*)   carve((size_t)N * DH * 4);

  const int B = 256;
  auto blks = [&](long n) { return (int)((n + B - 1) / B); };

  // --- normalization pre-pass ---
  k_zero     <<<blks(4L * Npad), B, 0, stream>>>(stats, 4 * Npad);
  k_degrees  <<<blks(E), B, 0, stream>>>(src, dst, ew, ow, iw, od, id, E);
  k_node_norm<<<blks(N), B, 0, stream>>>(ow, iw, od, id, N);
  k_edge_norm<<<blks(E), B, 0, stream>>>(src, dst, ew, ow, iw, nw, E);

  const float* layerW[3] = {W0, W1, W2};
  const float* layerB[3] = {b0, b1, b2};
  const int    layerD[3] = {DH, DH, NCLS};
  const float* h = features;

  for (int L = 0; L < 3; ++L) {
    const int dout = layerD[L];
    const int relu = (L < 2);
    float* out = (L == 2) ? (float*)d_out : Y;  // Y doubles as activation buffer

    k_scale_cvt    <<<blks((long)N * DH), B, 0, stream>>>(h, od, Xh, N * DH);
    k_transpose_cvt<<<blks((long)DH * dout), B, 0, stream>>>(layerW[L], WhT, dout);
    if (dout % 32 == 0) {             // 2 N-tiles per wave: A registers reused
      dim3 grid(Mpad / 16, dout / 32);
      k_gemm_wmma<2><<<grid, 32, 0, stream>>>(Xh, WhT, Y, dout);
    } else {                          // dout == 16 (classifier layer)
      dim3 grid(Mpad / 16, dout / 16);
      k_gemm_wmma<1><<<grid, 32, 0, stream>>>(Xh, WhT, Y, dout);
    }
    k_zero     <<<blks((long)N * dout), B, 0, stream>>>(agg, N * dout);
    k_edge_agg <<<blks((long)E * 32), B, 0, stream>>>(Y, nw, src, dst, agg, E, dout);
    k_finalize <<<blks((long)N * dout), B, 0, stream>>>(agg, id, layerB[L], out, N, dout, relu);

    h = out;
  }
}
